// RoIHeads_65369402245174
// MI455X (gfx1250) — matrix operand order repaired and checked by reading, not compiled, not run on hardware
//
#include <hip/hip_runtime.h>
#include <hip/hip_bf16.h>
#include <math.h>

// ---------------------------------------------------------------- types
typedef __attribute__((ext_vector_type(16))) __bf16 v16bf;
typedef __attribute__((ext_vector_type(8)))  __bf16 v8bf;
typedef __attribute__((ext_vector_type(8)))  float  v8f;

union BF16x16 { v16bf v; v8bf h[2]; };

// truncating fp32->bf16 pair pack: one v_perm_b32
__device__ __forceinline__ unsigned int pack_bf16x2(float lo, float hi) {
    return __builtin_amdgcn_perm(__float_as_uint(hi), __float_as_uint(lo),
                                 0x07060302u);
}

// ---------------------------------------------------------------- config
#define BM 128
#define BN 128
#define BK 32
#define NPROP 1024          // B*P
#define REP   1024
#define INF   12544
#define NCAND 2048          // P*(C-1)
#define NDET  100
#define BBOX_CLIP 4.135166556742356f

// ================================================================ GEMM
// C[M,N] = act(A[M,K] @ W[K,N] + bias[N]);  fp32 in/out, bf16 WMMA inside.
// Block: 256 thr (8 waves). Tile 128x128; wave tile 32x64 (2x4 WMMA accums).
// Register prefetch + double-buffered LDS: ONE barrier per K-iteration.
__global__ __launch_bounds__(256)
void gemm_bias_relu(const float* __restrict__ A, const float* __restrict__ W,
                    const float* __restrict__ bias, float* __restrict__ C,
                    int M, int N, int K, int relu)
{
    __shared__ __align__(16) unsigned short Xs[2][BM * BK];   // [m][k] bf16
    __shared__ __align__(16) unsigned short Ws[2][BN * BK];   // transposed [n][k]

    const int tid  = threadIdx.x;
    const int lane = tid & 31;
    const int wid  = tid >> 5;       // 0..7
    const int wm   = wid >> 1;       // 0..3  (row group, 32 rows each)
    const int wn   = wid & 1;        // 0..1  (col group, 64 cols each)
    const int m0   = blockIdx.y * BM;
    const int n0   = blockIdx.x * BN;
    const int l15  = lane & 15;
    const int lhi  = lane >> 4;      // 0 or 1

    v8f acc[2][4];
    #pragma unroll
    for (int i = 0; i < 2; ++i)
        #pragma unroll
        for (int j = 0; j < 4; ++j)
            acc[i][j] = v8f{0.f,0.f,0.f,0.f,0.f,0.f,0.f,0.f};

    // prefetch registers: A 8 x float2, B 8 x (two k-adjacent floats)
    float2 aR[8];
    float  bR[16];

    // A tile: 128x32 = 2048 float2 pairs; pair p: row = p>>4, colpair = p&15
    // B tile:  32x128 = 2048 k-pairs;     pair p: n = p&127, kpair = p>>7
    #define LOAD_A(K0)                                                          \
        _Pragma("unroll")                                                       \
        for (int i = 0; i < 8; ++i) {                                           \
            int p = i * 256 + tid;                                              \
            int r = p >> 4, cp = p & 15;                                        \
            aR[i] = *reinterpret_cast<const float2*>(                           \
                A + (size_t)(m0 + r) * K + (K0) + 2 * cp);                      \
        }
    #define LOAD_B(K0)                                                          \
        _Pragma("unroll")                                                       \
        for (int i = 0; i < 8; ++i) {                                           \
            int p = i * 256 + tid;                                              \
            int n = p & 127, kp = p >> 7;                                       \
            const float* src = W + (size_t)((K0) + 2 * kp) * N + (n0 + n);      \
            bR[2 * i]     = src[0];                                             \
            bR[2 * i + 1] = src[N];                                             \
        }

    LOAD_A(0)
    LOAD_B(0)

    int pbuf = 0;
    for (int k0 = 0; k0 < K; k0 += BK) {
        // convert + store staged registers into LDS (packed b32 stores)
        unsigned short* xs = Xs[pbuf];
        unsigned short* ws = Ws[pbuf];
        #pragma unroll
        for (int i = 0; i < 8; ++i) {
            int p = i * 256 + tid;
            int r = p >> 4, cp = p & 15;
            *reinterpret_cast<unsigned int*>(&xs[r * BK + 2 * cp]) =
                pack_bf16x2(aR[i].x, aR[i].y);
        }
        #pragma unroll
        for (int i = 0; i < 8; ++i) {
            int p = i * 256 + tid;
            int n = p & 127, kp = p >> 7;
            *reinterpret_cast<unsigned int*>(&ws[n * BK + 2 * kp]) =
                pack_bf16x2(bR[2 * i], bR[2 * i + 1]);
        }
        __syncthreads();

        // issue next tile's global loads early (latency hidden behind WMMAs)
        if (k0 + BK < K) {
            LOAD_A(k0 + BK)
            LOAD_B(k0 + BK)
        }

        // A fragment: lane m=l15; halves 0..7 -> K=khalf..+7, 8..15 -> K=16+khalf..+7
        BF16x16 afr[2], bfr[4];
        #pragma unroll
        for (int i = 0; i < 2; ++i) {
            const unsigned short* base = &xs[(wm * 32 + i * 16 + l15) * BK + lhi * 8];
            afr[i].h[0] = *reinterpret_cast<const v8bf*>(base);
            afr[i].h[1] = *reinterpret_cast<const v8bf*>(base + 16);
        }
        // B fragment: lane n=l15; halves 0..7 -> K=khalf..+7, 8..15 -> K=16+khalf..+7
        #pragma unroll
        for (int j = 0; j < 4; ++j) {
            const unsigned short* base = &ws[(wn * 64 + j * 16 + l15) * BK + lhi * 8];
            bfr[j].h[0] = *reinterpret_cast<const v8bf*>(base);
            bfr[j].h[1] = *reinterpret_cast<const v8bf*>(base + 16);
        }

        #pragma unroll
        for (int i = 0; i < 2; ++i)
            #pragma unroll
            for (int j = 0; j < 4; ++j)
                acc[i][j] = __builtin_amdgcn_wmma_f32_16x16x32_bf16(
                    false, afr[i].v, false, bfr[j].v,
                    (short)0, acc[i][j], false, false);

        pbuf ^= 1;
    }

    // epilogue: C/D layout: VGPR r -> M = r + 8*lhi, N = l15
    #pragma unroll
    for (int i = 0; i < 2; ++i) {
        #pragma unroll
        for (int j = 0; j < 4; ++j) {
            int col = n0 + wn * 64 + j * 16 + l15;
            float bv = bias[col];
            #pragma unroll
            for (int r = 0; r < 8; ++r) {
                int row = m0 + wm * 32 + i * 16 + lhi * 8 + r;
                float v = acc[i][j][r] + bv;
                if (relu) v = fmaxf(v, 0.0f);
                C[(size_t)row * N + col] = v;
            }
        }
    }
    #undef LOAD_A
    #undef LOAD_B
}

// ================================================= predictor + decode + softmax
// One block per proposal row. Computes 5 logits + 20 deltas, softmax, decode, clip.
__global__ __launch_bounds__(256)
void head_decode(const float* __restrict__ H,
                 const float* __restrict__ Wc, const float* __restrict__ bc,
                 const float* __restrict__ Wb, const float* __restrict__ bb,
                 const float* __restrict__ proposals,
                 float* __restrict__ pb,   // [2, 2048, 4]
                 float* __restrict__ sc)   // [2, 2048]
{
    const int row = blockIdx.x;
    const int tid = threadIdx.x;
    float pc[5]  = {0.f,0.f,0.f,0.f,0.f};
    float pbx[20];
    #pragma unroll
    for (int o = 0; o < 20; ++o) pbx[o] = 0.f;

    const float* h = H + (size_t)row * REP;
    for (int k = tid; k < REP; k += 256) {
        float hv = h[k];
        const float* wc = Wc + (size_t)k * 5;
        #pragma unroll
        for (int o = 0; o < 5; ++o) pc[o] += hv * wc[o];
        const float* wb = Wb + (size_t)k * 20;
        #pragma unroll
        for (int o = 0; o < 20; ++o) pbx[o] += hv * wb[o];
    }

    __shared__ float red[256];
    __shared__ float outv[25];
    for (int o = 0; o < 25; ++o) {
        red[tid] = (o < 5) ? pc[o] : pbx[o - 5];
        __syncthreads();
        for (int s2 = 128; s2 > 0; s2 >>= 1) {
            if (tid < s2) red[tid] += red[tid + s2];
            __syncthreads();
        }
        if (tid == 0) outv[o] = red[0] + ((o < 5) ? bc[o] : bb[o - 5]);
        __syncthreads();
    }

    if (tid == 0) {
        float mx = outv[0];
        for (int c = 1; c < 5; ++c) mx = fmaxf(mx, outv[c]);
        float ex[5], se = 0.f;
        for (int c = 0; c < 5; ++c) { ex[c] = expf(outv[c] - mx); se += ex[c]; }

        float x1 = proposals[row*4+0], y1 = proposals[row*4+1];
        float x2 = proposals[row*4+2], y2 = proposals[row*4+3];
        float w  = x2 - x1, hg = y2 - y1;
        float cx = x1 + 0.5f * w, cy = y1 + 0.5f * hg;
        int b = row >> 9, p = row & 511;

        for (int c = 1; c < 5; ++c) {
            float dx = outv[5 + c*4 + 0] * 0.1f;
            float dy = outv[5 + c*4 + 1] * 0.1f;
            float dw = fminf(outv[5 + c*4 + 2] * 0.2f, BBOX_CLIP);
            float dh = fminf(outv[5 + c*4 + 3] * 0.2f, BBOX_CLIP);
            float pcx = dx * w + cx, pcy = dy * hg + cy;
            float pw = expf(dw) * w, ph = expf(dh) * hg;
            float bx1 = fminf(fmaxf(pcx - 0.5f * pw, 0.f), 800.f);
            float by1 = fminf(fmaxf(pcy - 0.5f * ph, 0.f), 800.f);
            float bx2 = fminf(fmaxf(pcx + 0.5f * pw, 0.f), 800.f);
            float by2 = fminf(fmaxf(pcy + 0.5f * ph, 0.f), 800.f);
            int m = p * 4 + (c - 1);
            float* dst = pb + ((size_t)b * NCAND + m) * 4;
            dst[0] = bx1; dst[1] = by1; dst[2] = bx2; dst[3] = by2;
            sc[(size_t)b * NCAND + m] = ex[c] / se;
        }
    }
}

// ======================================================= sort + NMS + top-k
// One block per image (1024 threads). Bitonic sort desc by score, sequential
// suppression (class-equality == 801-offset trick), prefix-rank top-100 scatter.
__global__ __launch_bounds__(1024)
void nms_topk(const float* __restrict__ pb, const float* __restrict__ sc,
              float* __restrict__ outB, float* __restrict__ outS, int* __restrict__ outL)
{
    const int b   = blockIdx.x;
    const int tid = threadIdx.x;
    __shared__ float         s[NCAND];
    __shared__ short         id[NCAND];
    __shared__ float         bx[NCAND][4];
    __shared__ unsigned char keep[NCAND];
    __shared__ int           pfx[NCAND];

    for (int i = tid; i < NCAND; i += 1024) {
        s[i]  = sc[(size_t)b * NCAND + i];
        id[i] = (short)i;
    }
    __syncthreads();

    // bitonic sort, descending by score (payload = original index)
    for (int k = 2; k <= NCAND; k <<= 1) {
        for (int j = k >> 1; j > 0; j >>= 1) {
            for (int i = tid; i < NCAND; i += 1024) {
                int ixj = i ^ j;
                if (ixj > i) {
                    bool descSeg = ((i & k) == 0);
                    bool swap = descSeg ? (s[i] < s[ixj]) : (s[i] > s[ixj]);
                    if (swap) {
                        float ts = s[i]; s[i] = s[ixj]; s[ixj] = ts;
                        short ti = id[i]; id[i] = id[ixj]; id[ixj] = ti;
                    }
                }
            }
            __syncthreads();
        }
    }

    // gather sorted boxes, validity mask
    for (int i = tid; i < NCAND; i += 1024) {
        int o = id[i];
        const float* src = pb + ((size_t)b * NCAND + o) * 4;
        float x1 = src[0], y1 = src[1], x2 = src[2], y2 = src[3];
        bx[i][0] = x1; bx[i][1] = y1; bx[i][2] = x2; bx[i][3] = y2;
        keep[i] = (s[i] > 0.05f) && ((x2 - x1) >= 0.01f) && ((y2 - y1) >= 0.01f);
    }
    __syncthreads();

    // sequential greedy suppression (uniform branch on keep[i])
    for (int i = 0; i < NCAND - 1; ++i) {
        if (keep[i]) {
            float ax1 = bx[i][0], ay1 = bx[i][1], ax2 = bx[i][2], ay2 = bx[i][3];
            float aArea = (ax2 - ax1) * (ay2 - ay1);
            int la = id[i] & 3;
            for (int jj = i + 1 + tid; jj < NCAND; jj += 1024) {
                if (!keep[jj] || ((id[jj] & 3) != la)) continue;
                float ix1 = fmaxf(ax1, bx[jj][0]);
                float iy1 = fmaxf(ay1, bx[jj][1]);
                float ix2 = fminf(ax2, bx[jj][2]);
                float iy2 = fminf(ay2, bx[jj][3]);
                float iw = fmaxf(ix2 - ix1, 0.f), ih = fmaxf(iy2 - iy1, 0.f);
                float inter = iw * ih;
                float bArea = (bx[jj][2] - bx[jj][0]) * (bx[jj][3] - bx[jj][1]);
                float iou = inter / (aArea + bArea - inter + 1e-9f);
                if (iou > 0.5f) keep[jj] = 0;
            }
        }
        __syncthreads();
    }

    // exclusive prefix ranks (tiny; single-thread scan)
    if (tid == 0) {
        int run = 0;
        for (int i = 0; i < NCAND; ++i) { pfx[i] = run; run += keep[i] ? 1 : 0; }
    }
    __syncthreads();

    // zero outputs, then scatter top-100 in score order
    for (int q = tid; q < NDET; q += 1024) {
        float* db = outB + ((size_t)b * NDET + q) * 4;
        db[0] = db[1] = db[2] = db[3] = 0.f;
        outS[(size_t)b * NDET + q] = 0.f;
        outL[(size_t)b * NDET + q] = 0;
    }
    __syncthreads();
    for (int i = tid; i < NCAND; i += 1024) {
        if (keep[i] && pfx[i] < NDET) {
            int q = pfx[i];
            float* db = outB + ((size_t)b * NDET + q) * 4;
            db[0] = bx[i][0]; db[1] = bx[i][1]; db[2] = bx[i][2]; db[3] = bx[i][3];
            outS[(size_t)b * NDET + q] = s[i];
            outL[(size_t)b * NDET + q] = 1 + (id[i] & 3);
        }
    }
}

// ================================================================ launch
extern "C" void kernel_launch(void* const* d_in, const int* in_sizes, int n_in,
                              void* d_out, int out_size, void* d_ws, size_t ws_size,
                              hipStream_t stream)
{
    const float* box_features = (const float*)d_in[0]; // [1024, 12544]
    const float* proposals    = (const float*)d_in[1]; // [2, 512, 4]
    const float* W1 = (const float*)d_in[2];
    const float* bias1 = (const float*)d_in[3];
    const float* W2 = (const float*)d_in[4];
    const float* bias2 = (const float*)d_in[5];
    const float* Wc = (const float*)d_in[6];
    const float* bc = (const float*)d_in[7];
    const float* Wb = (const float*)d_in[8];
    const float* bb = (const float*)d_in[9];

    float* H1  = (float*)d_ws;                    // 1024*1024
    float* H2  = H1 + (size_t)NPROP * REP;        // 1024*1024
    float* pb  = H2 + (size_t)NPROP * REP;        // 2*2048*4
    float* scb = pb + (size_t)2 * NCAND * 4;      // 2*2048

    dim3 grid1(REP / BN, NPROP / BM);             // 8 x 8
    gemm_bias_relu<<<grid1, 256, 0, stream>>>(box_features, W1, bias1, H1,
                                              NPROP, REP, INF, 1);
    gemm_bias_relu<<<grid1, 256, 0, stream>>>(H1, W2, bias2, H2,
                                              NPROP, REP, REP, 1);
    head_decode<<<NPROP, 256, 0, stream>>>(H2, Wc, bc, Wb, bb, proposals, pb, scb);

    float* outBx = (float*)d_out;                 // [2,100,4]
    float* outS  = outBx + 2 * NDET * 4;          // [2,100]
    int*   outL  = (int*)(outS + 2 * NDET);       // [2,100]
    nms_topk<<<2, 1024, 0, stream>>>(pb, scb, outBx, outS, outL);
}
